// GCNLayer_56281251446800
// MI455X (gfx1250) — matrix operand order repaired and checked by reading, not compiled, not run on hardware
//
#include <hip/hip_runtime.h>

#define NNODE 8192
#define CIN   256
#define COUT  256

typedef __attribute__((ext_vector_type(16))) __bf16 v16bf;
typedef __attribute__((ext_vector_type(8)))  float  v8f;

// GEMM tiling: block = 256 threads = 8 waves; wave(wr,wc) owns 16 rows x 64 cols
#define BM 64
#define BN 128
#define BK 32
#define APITCH 40   // bf16 elems per LDS row (80B pitch: 16B aligned, conflict-free b128 reads)
#define BPITCH 40

union Frag32B {
  struct { uint4 lo; uint4 hi; } u;
  v16bf v;
};

static __device__ inline v16bf ld_frag16(const __bf16* p_lo, const __bf16* p_hi) {
  Frag32B f;
  f.u.lo = *reinterpret_cast<const uint4*>(p_lo);
  f.u.hi = *reinterpret_cast<const uint4*>(p_hi);
  return f.v;
}

// ---------------- small helper kernels ----------------

__global__ void cvt_bf16_kernel(const float* __restrict__ in, __bf16* __restrict__ out, int n) {
  int i = blockIdx.x * 256 + threadIdx.x;
  if (i < n) out[i] = (__bf16)in[i];
}

__global__ void deg_kernel(const float* __restrict__ adj, float* __restrict__ dinv) {
  __shared__ float red[256];
  const int i = blockIdx.x;
  const float* row = adj + (size_t)i * NNODE;
  float s = 0.f;
  for (int j = threadIdx.x; j < NNODE; j += 256) s += row[j];
  red[threadIdx.x] = s;
  __syncthreads();
  for (int off = 128; off > 0; off >>= 1) {
    if (threadIdx.x < off) red[threadIdx.x] += red[threadIdx.x + off];
    __syncthreads();
  }
  if (threadIdx.x == 0) {
    float d = red[0] + 1.0f;                    // self loop (A + I)
    dinv[i] = (d > 0.f) ? rsqrtf(d) : 0.f;
  }
}

__global__ void stats1_kernel(const float* __restrict__ X,
                              float* __restrict__ psum, float* __restrict__ psumsq) {
  const int c = threadIdx.x;
  const int i0 = blockIdx.x * 32;
  float s = 0.f, s2 = 0.f;
  for (int r = 0; r < 32; ++r) {
    float v = X[(size_t)(i0 + r) * COUT + c];
    s += v; s2 += v * v;
  }
  psum[blockIdx.x * COUT + c]   = s;
  psumsq[blockIdx.x * COUT + c] = s2;
}

__global__ void stats2_kernel(const float* __restrict__ psum, const float* __restrict__ psumsq,
                              float* __restrict__ sums, float* __restrict__ sumsq) {
  const int c = threadIdx.x;
  float s = 0.f, s2 = 0.f;
  for (int b = 0; b < NNODE / 32; ++b) {
    s  += psum[b * COUT + c];
    s2 += psumsq[b * COUT + c];
  }
  sums[c] = s; sumsq[c] = s2;
}

// Normalize + scale + dual-layout emit:
//   Y [i][c] = bf16( dinv[i] * (X[i][c]-mean[c]) / (std[c]+1e-6) )   (row-major, for epilogue)
//   Yt[c][i] = same value, channel-major (for coalesced b128 B staging in agg GEMM)
// 64x64 tile per block, LDS transpose. grid = (NNODE/64, COUT/64)
#define TP 72   // LDS transpose pitch in bf16 (144B: 16B aligned)
__global__ __launch_bounds__(256) void norm_t_kernel(const float* __restrict__ X,
                                                     const float* __restrict__ sums,
                                                     const float* __restrict__ sumsq,
                                                     const float* __restrict__ dinv,
                                                     __bf16* __restrict__ Y,
                                                     __bf16* __restrict__ Yt) {
  __shared__ float smean[64];
  __shared__ float sinv[64];
  __shared__ __bf16 T[64 * TP];
  const int tid = threadIdx.x;
  const int i0 = blockIdx.x * 64;
  const int c0 = blockIdx.y * 64;

  if (tid < 64) {
    const float invN = 1.0f / (float)NNODE;
    float mean = sums[c0 + tid] * invN;
    float var  = (sumsq[c0 + tid] - (float)NNODE * mean * mean) * (1.0f / (float)(NNODE - 1));
    float sd   = sqrtf(fmaxf(var, 0.f)) + 1e-6f;  // ddof=1, then +eps
    smean[tid] = mean;
    sinv[tid]  = 1.0f / sd;
  }
  __syncthreads();

  {   // phase 1: read X rows coalesced, normalize, write Y row-major + LDS transposed
    const int r = tid >> 2;          // 0..63 node row in tile
    const int q = tid & 3;           // 16-channel group
    const float dv = dinv[i0 + r];
    const float* xr = X + (size_t)(i0 + r) * COUT + c0 + q * 16;
    __bf16 vals[16];
#pragma unroll
    for (int f = 0; f < 4; ++f) {
      float4 v = reinterpret_cast<const float4*>(xr)[f];
      int cj = q * 16 + f * 4;
      vals[f * 4 + 0] = (__bf16)(dv * (v.x - smean[cj + 0]) * sinv[cj + 0]);
      vals[f * 4 + 1] = (__bf16)(dv * (v.y - smean[cj + 1]) * sinv[cj + 1]);
      vals[f * 4 + 2] = (__bf16)(dv * (v.z - smean[cj + 2]) * sinv[cj + 2]);
      vals[f * 4 + 3] = (__bf16)(dv * (v.w - smean[cj + 3]) * sinv[cj + 3]);
    }
    // row-major Y: 32B contiguous
    __bf16* yr = Y + (size_t)(i0 + r) * COUT + c0 + q * 16;
    reinterpret_cast<uint4*>(yr)[0] = *reinterpret_cast<uint4*>(&vals[0]);
    reinterpret_cast<uint4*>(yr)[1] = *reinterpret_cast<uint4*>(&vals[8]);
    // LDS transposed: T[c][r]
#pragma unroll
    for (int j = 0; j < 16; ++j) T[(q * 16 + j) * TP + r] = vals[j];
  }
  __syncthreads();
  {   // phase 2: write Yt channel-major, coalesced
    const int c = tid >> 2;          // 0..63 channel in tile
    const int q = tid & 3;           // 16-node group
    const __bf16* src = &T[c * TP + q * 16];
    __bf16* dst = Yt + (size_t)(c0 + c) * NNODE + i0 + q * 16;
    reinterpret_cast<uint4*>(dst)[0] = reinterpret_cast<const uint4*>(src)[0];
    reinterpret_cast<uint4*>(dst)[1] = reinterpret_cast<const uint4*>(src)[1];
  }
}

// ---------------- projection GEMM: X = nf_bf16 @ W_bf16^T + b ----------------
// Double-buffered LDS, one barrier per K-step.

__global__ __launch_bounds__(256) void proj_gemm(const __bf16* __restrict__ Anf,
                                                 const __bf16* __restrict__ Wb,
                                                 const float* __restrict__ bias,
                                                 float* __restrict__ X) {
  __shared__ __bf16 Asm[2 * BM * APITCH];
  __shared__ __bf16 Bsm[2 * BN * BPITCH];
  const int tid   = threadIdx.x;
  const int lane  = tid & 31;
  const int wave  = tid >> 5;
  const int wr    = wave >> 1;
  const int wc    = wave & 1;
  const int lcol  = lane & 15;
  const int khalf = lane >> 4;
  const int i0 = blockIdx.x * BM;
  const int n0 = blockIdx.y * BN;

  v8f acc[4];
#pragma unroll
  for (int t = 0; t < 4; ++t)
#pragma unroll
    for (int e = 0; e < 8; ++e) acc[t][e] = 0.f;

  const int arow = wr * 16 + lcol;
  int buf = 0;

  for (int k0 = 0; k0 < CIN; k0 += BK, buf ^= 1) {
    __bf16* As = &Asm[buf * BM * APITCH];
    __bf16* Bs = &Bsm[buf * BN * BPITCH];
    {   // A tile 64x32 bf16: one uint4 per thread
      int r = tid >> 2, q = tid & 3;
      uint4 v = *reinterpret_cast<const uint4*>(Anf + (size_t)(i0 + r) * CIN + k0 + q * 8);
      *reinterpret_cast<uint4*>(&As[r * APITCH + q * 8]) = v;
    }
    {   // B tile 128x32 bf16: Bs[n][k] = Wb[n0+n][k0+k]
      int n = tid >> 1, h = tid & 1;
      const __bf16* src = Wb + (size_t)(n0 + n) * CIN + k0 + h * 16;
      uint4 v0 = *reinterpret_cast<const uint4*>(src);
      uint4 v1 = *reinterpret_cast<const uint4*>(src + 8);
      *reinterpret_cast<uint4*>(&Bs[n * BPITCH + h * 16]) = v0;
      *reinterpret_cast<uint4*>(&Bs[n * BPITCH + h * 16 + 8]) = v1;
    }
    __syncthreads();

    const __bf16* ap = &As[arow * APITCH + khalf * 8];
    v16bf af = ld_frag16(ap, ap + 16);
#pragma unroll
    for (int nt = 0; nt < 4; ++nt) {
      const __bf16* bp = &Bs[(wc * 64 + nt * 16 + lcol) * BPITCH + khalf * 8];
      v16bf bfr = ld_frag16(bp, bp + 16);
      acc[nt] = __builtin_amdgcn_wmma_f32_16x16x32_bf16(false, af, false, bfr,
                                                        (short)0, acc[nt], false, false);
    }
  }

#pragma unroll
  for (int nt = 0; nt < 4; ++nt) {
    int col = n0 + wc * 64 + nt * 16 + lcol;
    float bv = bias[col];
#pragma unroll
    for (int r = 0; r < 8; ++r) {
      int gi = i0 + wr * 16 + khalf * 8 + r;
      X[(size_t)gi * COUT + col] = acc[nt][r] + bv;
    }
  }
}

// ---------------- aggregation GEMM: out = dinv .* (A_bf16 @ Y + Y) ----------------
// A converted fp32->bf16 per tile (0/1 exact); B staged b128 from channel-major Yt.

__global__ __launch_bounds__(256) void agg_gemm(const float* __restrict__ adj,
                                                const __bf16* __restrict__ Yt,
                                                const __bf16* __restrict__ Y,
                                                const float* __restrict__ dinv,
                                                float* __restrict__ out) {
  __shared__ __bf16 Asm[2 * BM * APITCH];
  __shared__ __bf16 Bsm[2 * BN * BPITCH];
  const int tid   = threadIdx.x;
  const int lane  = tid & 31;
  const int wave  = tid >> 5;
  const int wr    = wave >> 1;
  const int wc    = wave & 1;
  const int lcol  = lane & 15;
  const int khalf = lane >> 4;
  const int i0 = blockIdx.x * BM;
  const int n0 = blockIdx.y * BN;

  v8f acc[4];
#pragma unroll
  for (int t = 0; t < 4; ++t)
#pragma unroll
    for (int e = 0; e < 8; ++e) acc[t][e] = 0.f;

  const int arow = wr * 16 + lcol;
  int buf = 0;

  for (int k0 = 0; k0 < NNODE; k0 += BK, buf ^= 1) {
    __bf16* As = &Asm[buf * BM * APITCH];
    __bf16* Bs = &Bsm[buf * BN * BPITCH];
    {   // A tile 64x32 from fp32 adjacency, convert to bf16
      int r = tid >> 3, q = tid & 7;
      // prefetch a future A tile toward L2 (global_prefetch_b8)
      if (k0 + 128 < NNODE)
        __builtin_prefetch(adj + (size_t)(i0 + r) * NNODE + (k0 + 128) + q * 4, 0, 0);
#pragma unroll
      for (int p = 0; p < 2; ++p, r += 32) {
        float4 v = *reinterpret_cast<const float4*>(adj + (size_t)(i0 + r) * NNODE + k0 + q * 4);
        union { __bf16 h[4]; uint2 u; } pk;
        pk.h[0] = (__bf16)v.x; pk.h[1] = (__bf16)v.y;
        pk.h[2] = (__bf16)v.z; pk.h[3] = (__bf16)v.w;
        *reinterpret_cast<uint2*>(&As[(r - p * 32 + p * 32) * APITCH + q * 4]) = pk.u; // r already advanced
      }
    }
    {   // B tile: Bs[n][k] = Yt[n0+n][k0+k] -- pure b128 copies
      int n = tid >> 1, h = tid & 1;
      const __bf16* src = Yt + (size_t)(n0 + n) * NNODE + k0 + h * 16;
      uint4 v0 = *reinterpret_cast<const uint4*>(src);
      uint4 v1 = *reinterpret_cast<const uint4*>(src + 8);
      *reinterpret_cast<uint4*>(&Bs[n * BPITCH + h * 16]) = v0;
      *reinterpret_cast<uint4*>(&Bs[n * BPITCH + h * 16 + 8]) = v1;
    }
    __syncthreads();

    const __bf16* ap = &As[arow * APITCH + khalf * 8];
    v16bf af = ld_frag16(ap, ap + 16);
#pragma unroll
    for (int nt = 0; nt < 4; ++nt) {
      const __bf16* bp = &Bs[(wc * 64 + nt * 16 + lcol) * BPITCH + khalf * 8];
      v16bf bfr = ld_frag16(bp, bp + 16);
      acc[nt] = __builtin_amdgcn_wmma_f32_16x16x32_bf16(false, af, false, bfr,
                                                        (short)0, acc[nt], false, false);
    }
  }

  // epilogue: out[i][c] = dinv[i] * (agg + y[i][c])  (adds the +I self-loop term)
  int rows[8]; float dv[8];
#pragma unroll
  for (int r = 0; r < 8; ++r) {
    rows[r] = i0 + wr * 16 + khalf * 8 + r;
    dv[r] = dinv[rows[r]];
  }
#pragma unroll
  for (int nt = 0; nt < 4; ++nt) {
    int col = n0 + wc * 64 + nt * 16 + lcol;
#pragma unroll
    for (int r = 0; r < 8; ++r) {
      size_t idx = (size_t)rows[r] * COUT + col;
      out[idx] = dv[r] * (acc[nt][r] + (float)Y[idx]);
    }
  }
}

// ---------------- launcher ----------------

extern "C" void kernel_launch(void* const* d_in, const int* in_sizes, int n_in,
                              void* d_out, int out_size, void* d_ws, size_t ws_size,
                              hipStream_t stream) {
  const float* nf  = (const float*)d_in[0];   // [8192,256]
  const float* adj = (const float*)d_in[1];   // [8192,8192]
  const float* W   = (const float*)d_in[2];   // [256,256]
  const float* b   = (const float*)d_in[3];   // [256]
  float* out = (float*)d_out;                 // [8192,256]

  char* ws = (char*)d_ws;
  float*  dinv   = (float*)(ws + 0);                          //  32 KB
  float*  psum   = (float*)(ws + 32768);                      // 256 KB
  float*  psumsq = (float*)(ws + 32768 + 262144);             // 256 KB
  float*  sums   = (float*)(ws + 557056);                     //   1 KB
  float*  sumsq  = (float*)(ws + 558080);                     //   1 KB
  float*  X      = (float*)(ws + 559104);                     //   8 MB
  __bf16* nfb    = (__bf16*)(ws + 559104 + 8388608);          //   4 MB
  __bf16* Wb     = (__bf16*)(ws + 559104 + 8388608 + 4194304);            // 128 KB
  __bf16* Yb     = (__bf16*)(ws + 559104 + 8388608 + 4194304 + 131072);   //   4 MB
  __bf16* Ytb    = (__bf16*)(ws + 559104 + 8388608 + 4194304 + 131072 + 4194304); // 4 MB

  cvt_bf16_kernel<<<(NNODE * CIN + 255) / 256, 256, 0, stream>>>(nf, nfb, NNODE * CIN);
  cvt_bf16_kernel<<<(COUT * CIN + 255) / 256, 256, 0, stream>>>(W, Wb, COUT * CIN);
  deg_kernel<<<NNODE, 256, 0, stream>>>(adj, dinv);
  proj_gemm<<<dim3(NNODE / BM, COUT / BN), 256, 0, stream>>>(nfb, Wb, b, X);
  stats1_kernel<<<NNODE / 32, 256, 0, stream>>>(X, psum, psumsq);
  stats2_kernel<<<1, 256, 0, stream>>>(psum, psumsq, sums, sumsq);
  norm_t_kernel<<<dim3(NNODE / 64, COUT / 64), 256, 0, stream>>>(X, sums, sumsq, dinv, Yb, Ytb);
  agg_gemm<<<dim3(NNODE / BM, COUT / BN), 256, 0, stream>>>(adj, Ytb, Yb, dinv, out);
}
